// MultiLevelSelfAttentionTransformer_75788992905582
// MI455X (gfx1250) — matrix operand order repaired and checked by reading, not compiled, not run on hardware
//
#include <hip/hip_runtime.h>
#include <math.h>

// ---------------- problem constants ----------------
#define B_   4
#define S_   1024
#define D_   256
#define H_   8
#define HD_  32
#define L_   2
#define MLP_ 1024
#define WIN_ 64
#define NT_  (B_ * S_)          // 4096 tokens

typedef _Float16 v16h __attribute__((ext_vector_type(16)));
typedef _Float16 h8   __attribute__((ext_vector_type(8)));
typedef float    v8f  __attribute__((ext_vector_type(8)));

__device__ __forceinline__ v8f wmma_f16(v16h a, v16h b, v8f c) {
  // D = A(16x32 f16) x B(32x16 f16) + C(16x16 f32)
  return __builtin_amdgcn_wmma_f32_16x16x32_f16(
      false, a, false, b, (short)0, c, false, false);
}

__device__ __forceinline__ v16h cat8(h8 lo, h8 hi) {
  return __builtin_shufflevector(lo, hi, 0, 1, 2, 3, 4, 5, 6, 7,
                                 8, 9, 10, 11, 12, 13, 14, 15);
}

// reductions across the 16 lanes of a half-wave (masks 1..8 stay in-half)
__device__ __forceinline__ float halfmax(float v) {
#pragma unroll
  for (int m = 1; m <= 8; m <<= 1) v = fmaxf(v, __shfl_xor(v, m, 32));
  return v;
}
__device__ __forceinline__ float halfsum(float v) {
#pragma unroll
  for (int m = 1; m <= 8; m <<= 1) v += __shfl_xor(v, m, 32);
  return v;
}
__device__ __forceinline__ float wavesum(float v) {
#pragma unroll
  for (int m = 1; m <= 16; m <<= 1) v += __shfl_xor(v, m, 32);
  return v;
}

__device__ __forceinline__ float gelu_exact(float x) {
  return 0.5f * x * (1.0f + erff(x * 0.70710678118654752f));
}

// ---------------- f32 -> f16 conversion (weights, one shot) ----------------
__global__ void cvt_h_kernel(const float* __restrict__ x,
                             _Float16* __restrict__ y, int n) {
  int i = blockIdx.x * 256 + threadIdx.x;
  if (i < n) y[i] = (_Float16)x[i];
}

// ---------------- positional encoding add (writes f32 + f16) ----------------
__global__ void posadd_kernel(const float* __restrict__ x,
                              const float* __restrict__ scale,
                              float* __restrict__ y,
                              _Float16* __restrict__ yh) {
  size_t i = (size_t)blockIdx.x * blockDim.x + threadIdx.x;
  int s = (int)((i / D_) % S_);
  float v = x[i] + (float)s * scale[0];
  y[i] = v;
  yh[i] = (_Float16)v;
}

// ------ WMMA GEMM: Y = X(f16) * W(f16)^T + bias (+gelu)(+res) --------------
// X: (N,K); W: (O,K); wave tile 32x64 (2 A-frags x 4 B-frags, 8 acc chains).
// Block = 4 waves -> block tile 128 rows x 64 cols.
__global__ __launch_bounds__(128)
void gemm_h_kernel(const _Float16* __restrict__ X, const _Float16* __restrict__ W,
                   const float* __restrict__ bias, const float* __restrict__ res,
                   float* __restrict__ Yf, _Float16* __restrict__ Yh,
                   int K, int O, int act) {
  const int lane = threadIdx.x & 31;
  const int wave = threadIdx.x >> 5;
  const int l16 = lane & 15;
  const int half = lane >> 4;
  const int m0 = blockIdx.x * 128 + wave * 32;
  const int n0 = blockIdx.y * 64;

  v8f zero = {};
  v8f acc[2][4];
#pragma unroll
  for (int i = 0; i < 2; ++i)
#pragma unroll
    for (int j = 0; j < 4; ++j) acc[i][j] = zero;

  const _Float16* xr0 = X + (size_t)(m0 + l16) * K;
  const _Float16* xr1 = xr0 + (size_t)16 * K;
  const _Float16* wr[4];
#pragma unroll
  for (int j = 0; j < 4; ++j) wr[j] = W + (size_t)(n0 + 16 * j + l16) * K;

  for (int k0 = 0; k0 < K; k0 += 32) {
    // A fragments: lane row = l16; halves e<8 -> k = 8*half+e ; e>=8 -> 16+8*half+(e-8)
    v16h a0 = cat8(*(const h8*)(xr0 + k0 + 8 * half),
                   *(const h8*)(xr0 + k0 + 16 + 8 * half));
    v16h a1 = cat8(*(const h8*)(xr1 + k0 + 8 * half),
                   *(const h8*)(xr1 + k0 + 16 + 8 * half));
    // B fragments: lane col = l16; e<8 -> k = 8*half+e ; e>=8 -> 16+8*half+(e-8)
#pragma unroll
    for (int j = 0; j < 4; ++j) {
      v16h b = cat8(*(const h8*)(wr[j] + k0 + 8 * half),
                    *(const h8*)(wr[j] + k0 + 16 + 8 * half));
      acc[0][j] = wmma_f16(a0, b, acc[0][j]);
      acc[1][j] = wmma_f16(a1, b, acc[1][j]);
    }
    if (k0 + 32 < K) {
      __builtin_prefetch(xr0 + k0 + 32, 0, 3);
      __builtin_prefetch(wr[0] + k0 + 32, 0, 3);
    }
  }

  // C layout: lanes 0-15: (M=r, N=lane); lanes 16-31: (M=r+8, N=lane-16)
#pragma unroll
  for (int r = 0; r < 8; ++r) {
#pragma unroll
    for (int j = 0; j < 4; ++j) {
      int col = n0 + 16 * j + l16;
      float bval = bias[col];
#pragma unroll
      for (int i = 0; i < 2; ++i) {
        int row = m0 + 16 * i + r + 8 * half;
        float v = acc[i][j][r] + bval;
        if (act == 1) v = gelu_exact(v);
        if (res) v += res[(size_t)row * O + col];
        size_t idx = (size_t)row * O + col;
        if (Yf) Yf[idx] = v;
        if (Yh) Yh[idx] = (_Float16)v;
      }
    }
  }
}

// ---------------- fused flash-attention (one wave per (b,h,q-tile)) --------
// qkv (f16): (B, S, 3*D); Q at +0, K at +D, V at +2D; head h at +h*HD.
// Output: f16 (B*S, D) for the projection GEMM.
__global__ __launch_bounds__(32)
void attn_wmma_kernel(const _Float16* __restrict__ qkv,
                      _Float16* __restrict__ out, int banded) {
  __shared__ unsigned short Plds[16 * 32];   // P tile, f16 bits

  const int lane = threadIdx.x & 31;
  const int l16 = lane & 15;
  const int half = lane >> 4;
  const int q0 = blockIdx.x * 16;
  const int bh = blockIdx.y;
  const int b = bh / H_;
  const int h = bh % H_;
  const size_t base = (size_t)b * S_ * (3 * D_);
  const float rscale = 0.17677669529663687f;  // 1/sqrt(HD)

  // Q A-fragment (16 queries x HD=32)
  v16h qa;
  {
    const _Float16* qp = qkv + base + (size_t)(q0 + l16) * (3 * D_) + h * HD_;
    qa = cat8(*(const h8*)(qp + 8 * half), *(const h8*)(qp + 16 + 8 * half));
  }

  float mi[8], li[8];
  v8f O0 = {}, O1 = {};
#pragma unroll
  for (int r = 0; r < 8; ++r) { mi[r] = -1e9f; li[r] = 0.0f; }

  int kstart = 0, kend = S_;
  if (banded) {
    int ks = q0 - WIN_; if (ks < 0) ks = 0;
    kstart = ks & ~31;
    kend = q0 + 16 + WIN_; if (kend > S_) kend = S_;
  }

  for (int c0 = kstart; c0 < kend; c0 += 32) {
    // K^T B-fragments: col n = key; e<8 -> d = 8*half+e ; e>=8 -> 16+8*half+(e-8)
    const _Float16* kp0 = qkv + base + (size_t)(c0 + l16) * (3 * D_) + D_ + h * HD_;
    const _Float16* kp1 = kp0 + (size_t)16 * (3 * D_);
    v16h kb0 = cat8(*(const h8*)(kp0 + 8 * half), *(const h8*)(kp0 + 16 + 8 * half));
    v16h kb1 = cat8(*(const h8*)(kp1 + 8 * half), *(const h8*)(kp1 + 16 + 8 * half));
    v8f s0 = wmma_f16(qa, kb0, (v8f){});
    v8f s1 = wmma_f16(qa, kb1, (v8f){});

    float mnew[8];
#pragma unroll
    for (int r = 0; r < 8; ++r) {
      int row = q0 + r + 8 * half;
      int j0 = c0 + l16;
      int j1 = j0 + 16;
      float v0 = s0[r] * rscale;
      float v1 = s1[r] * rscale;
      if (banded) {
        if (j0 < row - WIN_ || j0 > row + WIN_) v0 = -1e9f;
        if (j1 < row - WIN_ || j1 > row + WIN_) v1 = -1e9f;
      }
      s0[r] = v0; s1[r] = v1;
      mnew[r] = fmaxf(mi[r], halfmax(fmaxf(v0, v1)));
    }
#pragma unroll
    for (int r = 0; r < 8; ++r) {
      float alpha = __expf(mi[r] - mnew[r]);
      float p0 = __expf(s0[r] - mnew[r]);
      float p1 = __expf(s1[r] - mnew[r]);
      li[r] = li[r] * alpha + halfsum(p0 + p1);
      O0[r] *= alpha;
      O1[r] *= alpha;
      mi[r] = mnew[r];
      int prow = r + 8 * half;
      Plds[prow * 32 + l16]      = __builtin_bit_cast(unsigned short, (_Float16)p0);
      Plds[prow * 32 + 16 + l16] = __builtin_bit_cast(unsigned short, (_Float16)p1);
    }
    asm volatile("s_wait_dscnt 0x0" ::: "memory");

    // reload P as A-fragment (row within lane)
    v16h pa;
#pragma unroll
    for (int i = 0; i < 8; ++i) {
      pa[i]     = __builtin_bit_cast(_Float16, Plds[l16 * 32 + 8 * half + i]);
      pa[8 + i] = __builtin_bit_cast(_Float16, Plds[l16 * 32 + 16 + 8 * half + i]);
    }
    // V B-fragments: col n = d; e<8 -> key = c0+8*half+e ; e>=8 -> c0+16+8*half+(e-8)
    v16h vb0, vb1;
#pragma unroll
    for (int i = 0; i < 16; ++i) {
      const _Float16* vrow = qkv + base +
          (size_t)(c0 + (i & 7) + 8 * half + 16 * (i >> 3)) * (3 * D_) + 2 * D_ + h * HD_;
      vb0[i] = vrow[l16];
      vb1[i] = vrow[16 + l16];
    }
    O0 = wmma_f16(pa, vb0, O0);
    O1 = wmma_f16(pa, vb1, O1);
  }

#pragma unroll
  for (int r = 0; r < 8; ++r) {
    int row = q0 + r + 8 * half;
    float inv = 1.0f / li[r];
    _Float16* op = out + ((size_t)b * S_ + row) * D_ + h * HD_;
    op[l16]      = (_Float16)(O0[r] * inv);
    op[16 + l16] = (_Float16)(O1[r] * inv);
  }
}

// -------- layernorm (one wave per token, D=256), writes f32 (+opt f16) -----
__global__ void ln_kernel(const float* __restrict__ x,
                          const float* __restrict__ g,
                          const float* __restrict__ bta,
                          float* __restrict__ y,
                          _Float16* __restrict__ yh) {
  int token = blockIdx.x * (blockDim.x >> 5) + (threadIdx.x >> 5);
  int lane = threadIdx.x & 31;
  const float* xp = x + (size_t)token * D_;
  float v[8];
  float s = 0.0f;
#pragma unroll
  for (int j = 0; j < 8; ++j) { v[j] = xp[lane + 32 * j]; s += v[j]; }
  s = wavesum(s);
  float mean = s * (1.0f / D_);
  float q = 0.0f;
#pragma unroll
  for (int j = 0; j < 8; ++j) { float d = v[j] - mean; q += d * d; }
  q = wavesum(q);
  float rstd = rsqrtf(q * (1.0f / D_) + 1e-5f);
#pragma unroll
  for (int j = 0; j < 8; ++j) {
    int d = lane + 32 * j;
    float o = (v[j] - mean) * rstd * g[d] + bta[d];
    y[(size_t)token * D_ + d] = o;
    if (yh) yh[(size_t)token * D_ + d] = (_Float16)o;
  }
}

// ---------------- host orchestration ----------------
static inline void launch_gemm(const _Float16* X, const _Float16* W,
                               const float* bias, const float* res,
                               float* Yf, _Float16* Yh, int K, int O, int act,
                               hipStream_t stream) {
  dim3 grid(NT_ / 128, O / 64);
  gemm_h_kernel<<<grid, 128, 0, stream>>>(X, W, bias, res, Yf, Yh, K, O, act);
}
static inline void launch_cvt(const float* x, _Float16* y, int n, hipStream_t stream) {
  cvt_h_kernel<<<(n + 255) / 256, 256, 0, stream>>>(x, y, n);
}

extern "C" void kernel_launch(void* const* d_in, const int* in_sizes, int n_in,
                              void* d_out, int out_size, void* d_ws, size_t ws_size,
                              hipStream_t stream) {
  const float* x      = (const float*)d_in[0];
  const float* scale  = (const float*)d_in[1];
  const float* Win_a  = (const float*)d_in[2];
  const float* bin_a  = (const float*)d_in[3];
  const float* Wout_a = (const float*)d_in[4];
  const float* bout_a = (const float*)d_in[5];
  const float* Win_e  = (const float*)d_in[6];
  const float* bin_e  = (const float*)d_in[7];
  const float* Wout_e = (const float*)d_in[8];
  const float* bout_e = (const float*)d_in[9];
  const float* ln1_g  = (const float*)d_in[10];
  const float* ln1_b  = (const float*)d_in[11];
  const float* ln2_g  = (const float*)d_in[12];
  const float* ln2_b  = (const float*)d_in[13];
  const float* W1     = (const float*)d_in[14];
  const float* b1     = (const float*)d_in[15];
  const float* W2     = (const float*)d_in[16];
  const float* b2     = (const float*)d_in[17];
  const float* lnf_g  = (const float*)d_in[18];
  const float* lnf_b  = (const float*)d_in[19];

  // ---- carve workspace (all chunks 256B-aligned by construction) ----
  char* p = (char*)d_ws;
  auto carve = [&](size_t bytes) {
    char* r = p;
    p += (bytes + 255) & ~(size_t)255;
    return r;
  };
  float*     Xf    = (float*)carve((size_t)NT_ * D_ * 4);
  float*     tmp   = (float*)carve((size_t)NT_ * D_ * 4);
  float*     attf  = (float*)carve((size_t)NT_ * D_ * 4);
  _Float16*  Xh    = (_Float16*)carve((size_t)NT_ * D_ * 2);
  _Float16*  qkvh  = (_Float16*)carve((size_t)NT_ * 3 * D_ * 2);
  _Float16*  attnh = (_Float16*)carve((size_t)NT_ * D_ * 2);   // attention out
  _Float16*  atth  = (_Float16*)carve((size_t)NT_ * D_ * 2);   // "attended"
  _Float16*  hh    = (_Float16*)carve((size_t)NT_ * MLP_ * 2); // MLP hidden
  _Float16*  WinaH = (_Float16*)carve((size_t)L_ * 3 * D_ * D_ * 2);
  _Float16*  WoutaH= (_Float16*)carve((size_t)L_ * D_ * D_ * 2);
  _Float16*  WineH = (_Float16*)carve((size_t)L_ * 3 * D_ * D_ * 2);
  _Float16*  WouteH= (_Float16*)carve((size_t)L_ * D_ * D_ * 2);
  _Float16*  W1H   = (_Float16*)carve((size_t)L_ * MLP_ * D_ * 2);
  _Float16*  W2H   = (_Float16*)carve((size_t)L_ * D_ * MLP_ * 2);

  // ---- one-shot weight conversion to f16 ----
  launch_cvt(Win_a,  WinaH,  L_ * 3 * D_ * D_, stream);
  launch_cvt(Wout_a, WoutaH, L_ * D_ * D_,     stream);
  launch_cvt(Win_e,  WineH,  L_ * 3 * D_ * D_, stream);
  launch_cvt(Wout_e, WouteH, L_ * D_ * D_,     stream);
  launch_cvt(W1,     W1H,    L_ * MLP_ * D_,   stream);
  launch_cvt(W2,     W2H,    L_ * D_ * MLP_,   stream);

  posadd_kernel<<<(NT_ * D_) / 256, 256, 0, stream>>>(x, scale, Xf, Xh);

  for (int l = 0; l < L_; ++l) {
    const _Float16* Wina  = WinaH  + (size_t)l * 3 * D_ * D_;
    const _Float16* Wouta = WoutaH + (size_t)l * D_ * D_;
    const _Float16* Wine  = WineH  + (size_t)l * 3 * D_ * D_;
    const _Float16* Woute = WouteH + (size_t)l * D_ * D_;
    const _Float16* W1l   = W1H + (size_t)l * MLP_ * D_;
    const _Float16* W2l   = W2H + (size_t)l * D_ * MLP_;
    const float* bina  = bin_a  + (size_t)l * 3 * D_;
    const float* bouta = bout_a + (size_t)l * D_;
    const float* bine  = bin_e  + (size_t)l * 3 * D_;
    const float* boute = bout_e + (size_t)l * D_;
    const float* b1l   = b1 + (size_t)l * MLP_;
    const float* b2l   = b2 + (size_t)l * D_;

    // --- unmasked MHA + LN1 ---
    launch_gemm(Xh, Wina, bina, nullptr, nullptr, qkvh, D_, 3 * D_, 0, stream);
    attn_wmma_kernel<<<dim3(S_ / 16, B_ * H_), 32, 0, stream>>>(qkvh, attnh, 0);
    launch_gemm(attnh, Wouta, bouta, Xf, tmp, nullptr, D_, D_, 0, stream);
    ln_kernel<<<NT_ / 8, 256, 0, stream>>>(tmp, ln1_g + l * D_, ln1_b + l * D_, Xf, Xh);

    // --- banded MHA + LN2 ---
    launch_gemm(Xh, Wine, bine, nullptr, nullptr, qkvh, D_, 3 * D_, 0, stream);
    attn_wmma_kernel<<<dim3(S_ / 16, B_ * H_), 32, 0, stream>>>(qkvh, attnh, 1);
    launch_gemm(attnh, Woute, boute, Xf, tmp, nullptr, D_, D_, 0, stream);
    ln_kernel<<<NT_ / 8, 256, 0, stream>>>(tmp, ln2_g + l * D_, ln2_b + l * D_, attf, atth);

    // --- MLP (attended in attf/atth) ---
    launch_gemm(atth, W1l, b1l, nullptr, nullptr, hh, D_, MLP_, 1, stream);   // gelu
    launch_gemm(hh, W2l, b2l, attf, Xf, Xh, MLP_, D_, 0, stream);             // +residual
  }

  ln_kernel<<<NT_ / 8, 256, 0, stream>>>(Xf, lnf_g, lnf_b, (float*)d_out, nullptr);
}
